// Fnn6_49864570306791
// MI455X (gfx1250) — matrix operand order repaired and checked by reading, not compiled, not run on hardware
//
#include <hip/hip_runtime.h>

// ---------------- problem constants ----------------
#define S_LEN   8192
#define CTX     21
#define EMB     128
#define GAZ     36
#define DRAW    (CTX * (EMB + GAZ))   // 3444
#define DPAD    3456                  // next multiple of 32
#define KT1     (DPAD / 32)           // 108 K-tiles for GEMM1
#define HID     1024
#define KT2     (HID / 32)            // 32 K-tiles for GEMM2
#define TAGS    32
#define START_TAG 30
#define STOP_TAG  31
#define NEGV   (-10000.0f)

typedef __attribute__((ext_vector_type(16))) __bf16 v16bf;
typedef __attribute__((ext_vector_type(8)))  float  v8f;

__device__ __forceinline__ unsigned short f2bf(float f) {
  union { float f; unsigned u; } v; v.f = f;
  unsigned u = v.u;
  u += 0x7FFFu + ((u >> 16) & 1u);   // round-to-nearest-even
  return (unsigned short)(u >> 16);
}

__device__ __forceinline__ v16bf ldfrag(const unsigned short* p) {
  return *(const v16bf*)p;
}

// ---------------------------------------------------------------------------
// Stage X = concat(emb[contexts], gazet) as bf16 in WMMA A-fragment order.
// A-tile (16x32 bf16): lane l holds M = l&15; element e of the v16bf maps to
//   k = ((e&8)?16:0) + (l>>4)*8 + (e&7)      (ISA 7.12.2, 16-bit A 16x32)
// Tile storage: tiles [mtile][ktile], 512 bf16 each, lane*16+e contiguous.
// ---------------------------------------------------------------------------
__global__ void k_stage_x(const int* __restrict__ contexts,
                          const float* __restrict__ gazet,
                          const float* __restrict__ emb,
                          unsigned short* __restrict__ Xbf) {
  int id   = blockIdx.x * blockDim.x + threadIdx.x;
  int w    = id & 511;
  int tile = id >> 9;
  int kt   = tile % KT1;
  int mt   = tile / KT1;
  if (mt >= S_LEN / 16) return;
  int lane = w >> 4;
  int e    = w & 15;
  int m    = (mt << 4) | (lane & 15);
  int k    = (kt << 5) + ((e & 8) ? 16 : 0) + ((lane >> 4) << 3) + (e & 7);
  float val = 0.0f;
  if (k < DRAW) {
    int c  = k / (EMB + GAZ);
    int r  = k - c * (EMB + GAZ);
    int sc = m * CTX + c;
    if (r < EMB) val = emb[contexts[sc] * EMB + r];
    else         val = gazet[sc * GAZ + (r - EMB)];
  }
  Xbf[id] = f2bf(val);
}

// ---------------------------------------------------------------------------
// Stage W1 (B = w1ᵀ) in B-fragment order. B-tile (32x16): lane l holds
// N = l&15; element e maps to k = ((e&8)?16:0) + (l>>4)*8 + (e&7). Tiles [ntile][ktile].
// ---------------------------------------------------------------------------
__global__ void k_stage_w1(const float* __restrict__ w1,
                           unsigned short* __restrict__ W1bf) {
  int id   = blockIdx.x * blockDim.x + threadIdx.x;
  int w    = id & 511;
  int tile = id >> 9;
  int kt   = tile % KT1;
  int nt   = tile / KT1;
  if (nt >= HID / 16) return;
  int lane = w >> 4;
  int e    = w & 15;
  int n    = (nt << 4) | (lane & 15);
  int k    = (kt << 5) + ((e & 8) ? 16 : 0) + ((lane >> 4) << 3) + (e & 7);
  float val = (k < DRAW) ? w1[n * DRAW + k] : 0.0f;
  W1bf[id] = f2bf(val);
}

__global__ void k_stage_w2(const float* __restrict__ w2,
                           unsigned short* __restrict__ W2bf) {
  int id   = blockIdx.x * blockDim.x + threadIdx.x;
  if (id >= 2 * KT2 * 512) return;
  int w    = id & 511;
  int tile = id >> 9;
  int kt   = tile % KT2;
  int nt   = tile / KT2;
  int lane = w >> 4;
  int e    = w & 15;
  int n    = (nt << 4) | (lane & 15);
  int k    = (kt << 5) + ((e & 8) ? 16 : 0) + ((lane >> 4) << 3) + (e & 7);
  W2bf[id] = f2bf(w2[n * HID + k]);
}

// ---------------------------------------------------------------------------
// GEMM1 epilogue: bias + ReLU, emit bf16 directly into GEMM2's A-fragment
// layout (K-dim of GEMM2 = hidden index).
// C/D layout (ISA): element (r, lane) -> m = r + 8*(lane>>4), n = lane&15.
// ---------------------------------------------------------------------------
__device__ __forceinline__ void gemm1_epilogue(v8f a, int mt, int ntile, int lane,
                                               const float* __restrict__ b1,
                                               unsigned short* __restrict__ Hbf) {
  int hl = lane >> 4, nloc = lane & 15;
  int n_g = ntile * 16 + nloc;           // hidden unit = K index for GEMM2
  float bias = b1[n_g];
  int kt2 = n_g >> 5;
  int kl  = n_g & 31;
  int hl2 = (kl >> 3) & 1;
  int e2  = (kl & 7) | ((kl & 16) ? 8 : 0);
#pragma unroll
  for (int r = 0; r < 8; ++r) {
    int m_loc = r + (hl << 3);
    float x = fmaxf(a[r] + bias, 0.0f);
    int lane2 = m_loc | (hl2 << 4);
    Hbf[((size_t)mt * KT2 + kt2) * 512 + lane2 * 16 + e2] = f2bf(x);
  }
}

// ---------------------------------------------------------------------------
// GEMM1: h = relu(X @ W1ᵀ + b1). One wave owns a 32x64 output tile
// (2 A-frags x 4 B-frags -> 8 WMMAs per K-tile), explicitly double-buffered
// so next K-tile's 6 fragment loads are in flight during the 8 WMMAs.
// ---------------------------------------------------------------------------
__global__ void __launch_bounds__(256)
k_gemm1(const unsigned short* __restrict__ Xbf,
        const unsigned short* __restrict__ W1bf,
        const float* __restrict__ b1,
        unsigned short* __restrict__ Hbf) {
  const int lane = threadIdx.x & 31;
  const int wid  = threadIdx.x >> 5;
  const int mt2  = blockIdx.x * 8 + wid;   // 0..255 : 32-row supertile
  const int nt0  = blockIdx.y * 4;         // 0..63 step 4
  const int mtA  = mt2 * 2;
  const int mtB  = mtA + 1;

  v8f z = {0.f,0.f,0.f,0.f,0.f,0.f,0.f,0.f};
  v8f acc00 = z, acc01 = z, acc02 = z, acc03 = z;
  v8f acc10 = z, acc11 = z, acc12 = z, acc13 = z;

  const unsigned short* Ab0 = Xbf  + (size_t)mtA * KT1 * 512 + lane * 16;
  const unsigned short* Ab1 = Xbf  + (size_t)mtB * KT1 * 512 + lane * 16;
  const unsigned short* Bb  = W1bf + (size_t)nt0 * KT1 * 512 + lane * 16;

  v16bf a0 = ldfrag(Ab0);
  v16bf a1 = ldfrag(Ab1);
  v16bf b0 = ldfrag(Bb);
  v16bf b1v = ldfrag(Bb + (size_t)KT1 * 512);
  v16bf b2v = ldfrag(Bb + (size_t)(2 * KT1) * 512);
  v16bf b3v = ldfrag(Bb + (size_t)(3 * KT1) * 512);

  for (int kt = 0; kt < KT1; ++kt) {
    int ktn = (kt + 1 < KT1) ? kt + 1 : kt;     // clamp: last iter reloads self
    v16bf na0 = ldfrag(Ab0 + (size_t)ktn * 512);
    v16bf na1 = ldfrag(Ab1 + (size_t)ktn * 512);
    v16bf nb0 = ldfrag(Bb + (size_t)ktn * 512);
    v16bf nb1 = ldfrag(Bb + (size_t)(KT1     + ktn) * 512);
    v16bf nb2 = ldfrag(Bb + (size_t)(2 * KT1 + ktn) * 512);
    v16bf nb3 = ldfrag(Bb + (size_t)(3 * KT1 + ktn) * 512);

    acc00 = __builtin_amdgcn_wmma_f32_16x16x32_bf16(false, a0, false, b0,  (short)0, acc00, false, false);
    acc01 = __builtin_amdgcn_wmma_f32_16x16x32_bf16(false, a0, false, b1v, (short)0, acc01, false, false);
    acc02 = __builtin_amdgcn_wmma_f32_16x16x32_bf16(false, a0, false, b2v, (short)0, acc02, false, false);
    acc03 = __builtin_amdgcn_wmma_f32_16x16x32_bf16(false, a0, false, b3v, (short)0, acc03, false, false);
    acc10 = __builtin_amdgcn_wmma_f32_16x16x32_bf16(false, a1, false, b0,  (short)0, acc10, false, false);
    acc11 = __builtin_amdgcn_wmma_f32_16x16x32_bf16(false, a1, false, b1v, (short)0, acc11, false, false);
    acc12 = __builtin_amdgcn_wmma_f32_16x16x32_bf16(false, a1, false, b2v, (short)0, acc12, false, false);
    acc13 = __builtin_amdgcn_wmma_f32_16x16x32_bf16(false, a1, false, b3v, (short)0, acc13, false, false);

    a0 = na0; a1 = na1; b0 = nb0; b1v = nb1; b2v = nb2; b3v = nb3;
  }

  gemm1_epilogue(acc00, mtA, nt0 + 0, lane, b1, Hbf);
  gemm1_epilogue(acc01, mtA, nt0 + 1, lane, b1, Hbf);
  gemm1_epilogue(acc02, mtA, nt0 + 2, lane, b1, Hbf);
  gemm1_epilogue(acc03, mtA, nt0 + 3, lane, b1, Hbf);
  gemm1_epilogue(acc10, mtB, nt0 + 0, lane, b1, Hbf);
  gemm1_epilogue(acc11, mtB, nt0 + 1, lane, b1, Hbf);
  gemm1_epilogue(acc12, mtB, nt0 + 2, lane, b1, Hbf);
  gemm1_epilogue(acc13, mtB, nt0 + 3, lane, b1, Hbf);
}

// ---------------------------------------------------------------------------
// GEMM2: feats = H @ W2ᵀ + b2. One wave per 32x32 output tile (2 A-frags x
// 2 B-frags -> 4 WMMAs per K-tile), double-buffered.
// ---------------------------------------------------------------------------
__global__ void __launch_bounds__(256)
k_gemm2(const unsigned short* __restrict__ Hbf,
        const unsigned short* __restrict__ W2bf,
        const float* __restrict__ b2,
        float* __restrict__ feats) {
  const int lane = threadIdx.x & 31;
  const int wid  = threadIdx.x >> 5;
  const int mt2  = blockIdx.x * 8 + wid;   // 0..255
  const int mtA  = mt2 * 2;
  const int mtB  = mtA + 1;

  v8f z = {0.f,0.f,0.f,0.f,0.f,0.f,0.f,0.f};
  v8f acc00 = z, acc01 = z, acc10 = z, acc11 = z;
  const unsigned short* Ab0 = Hbf + (size_t)mtA * KT2 * 512 + lane * 16;
  const unsigned short* Ab1 = Hbf + (size_t)mtB * KT2 * 512 + lane * 16;
  const unsigned short* Bb  = W2bf + lane * 16;

  v16bf a0 = ldfrag(Ab0);
  v16bf a1 = ldfrag(Ab1);
  v16bf b0 = ldfrag(Bb);
  v16bf b1v = ldfrag(Bb + (size_t)KT2 * 512);

  for (int kt = 0; kt < KT2; ++kt) {
    int ktn = (kt + 1 < KT2) ? kt + 1 : kt;
    v16bf na0 = ldfrag(Ab0 + (size_t)ktn * 512);
    v16bf na1 = ldfrag(Ab1 + (size_t)ktn * 512);
    v16bf nb0 = ldfrag(Bb + (size_t)ktn * 512);
    v16bf nb1 = ldfrag(Bb + (size_t)(KT2 + ktn) * 512);

    acc00 = __builtin_amdgcn_wmma_f32_16x16x32_bf16(false, a0, false, b0,  (short)0, acc00, false, false);
    acc01 = __builtin_amdgcn_wmma_f32_16x16x32_bf16(false, a0, false, b1v, (short)0, acc01, false, false);
    acc10 = __builtin_amdgcn_wmma_f32_16x16x32_bf16(false, a1, false, b0,  (short)0, acc10, false, false);
    acc11 = __builtin_amdgcn_wmma_f32_16x16x32_bf16(false, a1, false, b1v, (short)0, acc11, false, false);

    a0 = na0; a1 = na1; b0 = nb0; b1v = nb1;
  }

  int hl = lane >> 4, nloc = lane & 15;
  float biasLo = b2[nloc], biasHi = b2[16 + nloc];
#pragma unroll
  for (int r = 0; r < 8; ++r) {
    int mA = mtA * 16 + r + (hl << 3);
    int mB = mtB * 16 + r + (hl << 3);
    feats[(size_t)mA * TAGS + nloc]      = acc00[r] + biasLo;
    feats[(size_t)mA * TAGS + 16 + nloc] = acc01[r] + biasHi;
    feats[(size_t)mB * TAGS + nloc]      = acc10[r] + biasLo;
    feats[(size_t)mB * TAGS + 16 + nloc] = acc11[r] + biasHi;
  }
}

// ---------------------------------------------------------------------------
// CRF: single wave32. lane = "next" tag. exp(trans) resident in VGPRs.
// new_alpha[n] = feat[n] + m + log( sum_p exp(trans[n][p]) * exp(alpha[p]-m) )
// ---------------------------------------------------------------------------
__global__ void __launch_bounds__(32)
k_crf(const float* __restrict__ feats,
      const float* __restrict__ trans,
      const int* __restrict__ labels,
      float* __restrict__ out) {
  const int lane = threadIdx.x;

  float et[TAGS];
#pragma unroll 8
  for (int p = 0; p < TAGS; ++p) et[p] = __expf(trans[lane * TAGS + p]);

  // ---- gold score (parallel over sequence, then wave reduce) ----
  float gold = 0.0f;
  for (int i = lane; i < S_LEN; i += 32) {
    int tag  = labels[i];
    int prev = (i == 0) ? START_TAG : labels[i - 1];
    gold += feats[(size_t)i * TAGS + tag] + trans[tag * TAGS + prev];
  }
  if (lane == 0) gold += trans[STOP_TAG * TAGS + labels[S_LEN - 1]];
  for (int off = 16; off >= 1; off >>= 1) gold += __shfl_xor(gold, off, 32);

  // ---- forward scan ----
  float alpha = (lane == START_TAG) ? 0.0f : NEGV;
  float featv = feats[lane];
  for (int s = 0; s < S_LEN; ++s) {
    float featn = (s + 1 < S_LEN) ? feats[(size_t)(s + 1) * TAGS + lane] : 0.0f;
    if (s + 8 < S_LEN)
      __builtin_prefetch(&feats[(size_t)(s + 8) * TAGS + lane], 0, 0);
    float m = alpha;
    for (int off = 16; off >= 1; off >>= 1) m = fmaxf(m, __shfl_xor(m, off, 32));
    float ee = __expf(alpha - m);
    float ssum = 0.0f;
#pragma unroll
    for (int p = 0; p < TAGS; ++p) ssum = fmaf(et[p], __shfl(ee, p, 32), ssum);
    alpha = featv + m + __logf(ssum);
    featv = featn;
  }
  alpha += trans[STOP_TAG * TAGS + lane];
  float m = alpha;
  for (int off = 16; off >= 1; off >>= 1) m = fmaxf(m, __shfl_xor(m, off, 32));
  float ee = __expf(alpha - m);
  float ssum = ee;
  for (int off = 16; off >= 1; off >>= 1) ssum += __shfl_xor(ssum, off, 32);
  float fwd = m + __logf(ssum);
  if (lane == 0) out[0] = fwd - gold;
}

// ---------------------------------------------------------------------------
extern "C" void kernel_launch(void* const* d_in, const int* in_sizes, int n_in,
                              void* d_out, int out_size, void* d_ws, size_t ws_size,
                              hipStream_t stream) {
  const int*   contexts = (const int*)  d_in[0];
  const float* gazet    = (const float*)d_in[1];
  const int*   labels   = (const int*)  d_in[2];
  const float* emb      = (const float*)d_in[3];
  const float* w1       = (const float*)d_in[4];
  const float* b1       = (const float*)d_in[5];
  const float* w2       = (const float*)d_in[6];
  const float* b2       = (const float*)d_in[7];
  const float* trans    = (const float*)d_in[8];
  float* out = (float*)d_out;

  char* ws = (char*)d_ws;
  unsigned short* Xbf  = (unsigned short*)(ws);                // 8192*3456*2 = 56,623,104
  unsigned short* W1bf = (unsigned short*)(ws + 56623104);     // 1024*3456*2 =  7,077,888
  unsigned short* Hbf  = (unsigned short*)(ws + 63700992);     // 8192*1024*2 = 16,777,216
  unsigned short* W2bf = (unsigned short*)(ws + 80478208);     //   32*1024*2 =     65,536
  float*          feats= (float*)        (ws + 80543744);      // 8192*32*4   =  1,048,576

  {
    int total = (S_LEN / 16) * KT1 * 512;      // 28,311,552
    k_stage_x<<<total / 256, 256, 0, stream>>>(contexts, gazet, emb, Xbf);
  }
  {
    int total = (HID / 16) * KT1 * 512;        // 3,538,944
    k_stage_w1<<<total / 256, 256, 0, stream>>>(w1, W1bf);
  }
  {
    int total = 2 * KT2 * 512;                 // 32,768
    k_stage_w2<<<total / 256, 256, 0, stream>>>(w2, W2bf);
  }
  {
    dim3 g(32, 16);                            // 256 m-supertiles x 16 n-groups
    k_gemm1<<<g, 256, 0, stream>>>(Xbf, W1bf, b1, Hbf);
  }
  k_gemm2<<<32, 256, 0, stream>>>(Hbf, W2bf, b2, feats);
  k_crf<<<1, 32, 0, stream>>>(feats, trans, labels, out);
}
